// LinearAttention_30150670417974
// MI455X (gfx1250) — matrix operand order repaired and checked by reading, not compiled, not run on hardware
//
#include <hip/hip_runtime.h>
#include <hip/hip_bf16.h>

// ---------------------------------------------------------------------------
// LinearAttention for MI455X (gfx1250): bf16 WMMA pipeline, f32 accumulate.
//   K1: qkv = (x*(g+1)*16) @ Wqkv + b      -> q(*scale),k,v bf16 [bh][4096][64]
//   K2: ctx = k^T v + memk^T memv           -> bf16 ctxT [d2][d1] (transposed)
//   K3: out = q @ ctx                       -> bf16 [b][n][h][64] (= [M][512])
//   K4: y = out @ Wo + bo, * (go+1)*16      -> f32 [M][256]
// LDS tiles are staged so every WMMA fragment is two contiguous 16-byte
// per-lane reads (ds_load_b128). Pure bf16 tile copies (K3 Q/ctx, K4 A) go
// through the Tensor Data Mover (TENSOR_LOAD_TO_LDS) with LDS row padding,
// synchronized with s_wait_tensorcnt.
// ---------------------------------------------------------------------------

typedef __bf16 bf16;
typedef bf16  v16bf __attribute__((ext_vector_type(16)));
typedef bf16  v8bf  __attribute__((ext_vector_type(8)));
typedef bf16  v4bf  __attribute__((ext_vector_type(4)));
typedef bf16  v2bf  __attribute__((ext_vector_type(2)));
typedef float v8f   __attribute__((ext_vector_type(8)));
typedef unsigned int v4u __attribute__((ext_vector_type(4)));
typedef int   v8i   __attribute__((ext_vector_type(8)));
typedef int   v4i   __attribute__((ext_vector_type(4)));

#define BATCH   8
#define NTOK    4096            // 64*64 per image
#define MROWS   (BATCH * NTOK)  // 32768
#define DIM     256
#define HEADS   8
#define HD      64
#define HIDDEN  512
#define NQKV    1536
#define RMS_MUL 16.0f           // sqrt(256)
#define QSCALE  0.125f          // 64^-0.5

// ---------------- WMMA fragment helpers (ISA 7.12.2 layouts) ----------------

__device__ __forceinline__ v16bf cat8(v8bf lo, v8bf hi) {
  return __builtin_shufflevector(lo, hi, 0, 1, 2, 3, 4, 5, 6, 7,
                                 8, 9, 10, 11, 12, 13, 14, 15);
}

// A fragment 16x32 bf16 from row-major LDS tile (rows 16B-aligned).
__device__ __forceinline__ v16bf frag_a(const bf16* base, int ld, int mbase,
                                        int kbase, int lane) {
  const int row  = mbase + (lane & 15);
  const int koff = (lane & 16) ? 8 : 0;
  const bf16* p  = base + row * ld + kbase + koff;
  return cat8(*(const v8bf*)p, *(const v8bf*)(p + 16));
}

// B fragment 32x16 bf16 from *transposed* LDS tile BT[n][k].
__device__ __forceinline__ v16bf frag_bT(const bf16* base, int ld, int kbase,
                                         int nbase, int lane) {
  const bf16* p = base + (nbase + (lane & 15)) * ld + kbase +
                  ((lane & 16) ? 8 : 0);
  return cat8(*(const v8bf*)p, *(const v8bf*)(p + 16));
}

__device__ __forceinline__ v8f wmma_bf16(v16bf a, v16bf b, v8f c) {
  return __builtin_amdgcn_wmma_f32_16x16x32_bf16(false, a, false, b,
                                                 (short)0, c, false, false);
}

__device__ __forceinline__ void pack2(bf16* dst, bf16 lo, bf16 hi) {
  v2bf p;
  p[0] = lo;
  p[1] = hi;
  *(v2bf*)dst = p;
}

// ---------------- Tensor Data Mover: 2D bf16 tile -> padded LDS -------------
// D# layout per CDNA5 ISA. The low 32 bits of a generic
// LDS pointer are the LDS byte offset (hardware truncates addr[31:0], §10.2).
// pad_interval code: bytes-per-row = 8<<code; pad_amount code: (code+1) dwords.
// This toolchain exposes the 6-arg builtin: (g0, g1, g2, g3, v8i, cpol);
// groups 2/3 and the trailing group are unused for a 2-D tile (zeros).
__device__ __forceinline__ void tdm_load_2d_bf16(void* lds_dst,
                                                 const void* gsrc,
                                                 unsigned tensor_d0,
                                                 unsigned tensor_d1,
                                                 unsigned tile_d0,
                                                 unsigned tile_d1,
                                                 unsigned d0_stride,
                                                 unsigned pad_interval_code,
                                                 unsigned pad_amount_code) {
  const unsigned lds_addr = (unsigned)(unsigned long long)lds_dst;
  const unsigned long long ga = (unsigned long long)gsrc;
  v4u g0;
  g0[0] = 1u;                                           // count=1, user D#
  g0[1] = lds_addr;                                     // lds_addr
  g0[2] = (unsigned)(ga & 0xFFFFFFFFu);                 // global_addr[31:0]
  g0[3] = (unsigned)((ga >> 32) & 0x01FFFFFFu) | (2u << 30);  // [56:32]|type=2
  v8i g1;
  g1[0] = (int)((1u << 16)                              // data_size = 2B
                | (1u << 20)                            // pad_enable
                | (pad_interval_code << 22)
                | (pad_amount_code << 25));
  g1[1] = (int)((tensor_d0 & 0xFFFFu) << 16);           // tensor_dim0[15:0]
  g1[2] = (int)(((tensor_d0 >> 16) & 0xFFFFu)           // tensor_dim0[31:16]
                | ((tensor_d1 & 0xFFFFu) << 16));       // tensor_dim1[15:0]
  g1[3] = (int)(((tensor_d1 >> 16) & 0xFFFFu)           // tensor_dim1[31:16]
                | ((tile_d0 & 0xFFFFu) << 16));         // tile_dim0
  g1[4] = (int)(tile_d1 & 0xFFFFu);                     // tile_dim1, tile_dim2=0
  g1[5] = (int)d0_stride;                               // dim0_stride[31:0]
  g1[6] = 0;                                            // stride hi / dim1_stride
  g1[7] = 0;
  v4i z4 = {0, 0, 0, 0};                                // 2D: groups 2/3 unused
  v8i z8 = {0, 0, 0, 0, 0, 0, 0, 0};
  __builtin_amdgcn_tensor_load_to_lds(g0, g1, z4, z4, z8, 0);
}

// Stage a 128x64 f32 weight sub-block transposed into BT[64][136] as bf16.
__device__ __forceinline__ void stage_wT(bf16* BT, const float* W, int ldw,
                                         int kb, int nb, int t) {
  const int kp2 = (t & 63) * 2;
  const int ng  = (t >> 6) * 16;
#pragma unroll
  for (int v = 0; v < 4; ++v) {
    float4 f0 = *(const float4*)&W[(size_t)(kb + kp2) * ldw + nb + ng + v * 4];
    float4 f1 = *(const float4*)&W[(size_t)(kb + kp2 + 1) * ldw + nb + ng + v * 4];
    pack2(&BT[(ng + v * 4 + 0) * 136 + kp2], (bf16)f0.x, (bf16)f1.x);
    pack2(&BT[(ng + v * 4 + 1) * 136 + kp2], (bf16)f0.y, (bf16)f1.y);
    pack2(&BT[(ng + v * 4 + 2) * 136 + kp2], (bf16)f0.z, (bf16)f1.z);
    pack2(&BT[(ng + v * 4 + 3) * 136 + kp2], (bf16)f0.w, (bf16)f1.w);
  }
}

// ---------------- K1: fused norm + QKV GEMM ---------------------------------
// grid(24, 1024) block 256: block tile 32(M) x 64(N), K staged in 128-chunks.
__global__ __launch_bounds__(256)
void k1_qkv(const float* __restrict__ x, const float* __restrict__ gamma,
            const float* __restrict__ W, const float* __restrict__ bias,
            bf16* __restrict__ q_ws, bf16* __restrict__ k_ws,
            bf16* __restrict__ v_ws) {
  __shared__ __align__(16) bf16 As[32][136];     // row-major  [m][k]
  __shared__ __align__(16) bf16 BT[64][136];     // transposed [n][k]
  const int t = threadIdx.x, lane = t & 31, wave = t >> 5;
  const int nb = blockIdx.x * 64;
  const int mb = blockIdx.y * 32;
  const int mbase = (wave >> 2) * 16;     // 0 / 16
  const int nbase = (wave & 3) * 16;      // 0..48
  v8f acc = {};

#pragma unroll
  for (int kc = 0; kc < 2; ++kc) {
    const int kb = kc * 128;
    {   // stage A: 32 x 128 f32 -> bf16 with (gamma+1)*16 scaling
      const int row = t >> 3, seg = (t & 7) * 16;
#pragma unroll
      for (int v = 0; v < 4; ++v) {
        const int c = kb + seg + v * 4;
        float4 f = *(const float4*)&x[(size_t)(mb + row) * DIM + c];
        As[row][seg + v * 4 + 0] = (bf16)(f.x * (gamma[c + 0] + 1.0f) * RMS_MUL);
        As[row][seg + v * 4 + 1] = (bf16)(f.y * (gamma[c + 1] + 1.0f) * RMS_MUL);
        As[row][seg + v * 4 + 2] = (bf16)(f.z * (gamma[c + 2] + 1.0f) * RMS_MUL);
        As[row][seg + v * 4 + 3] = (bf16)(f.w * (gamma[c + 3] + 1.0f) * RMS_MUL);
      }
    }
    stage_wT(&BT[0][0], W, NQKV, kb, nb, t);
    __syncthreads();
#pragma unroll
    for (int ks = 0; ks < 128; ks += 32) {
      v16bf a = frag_a(&As[0][0], 136, mbase, ks, lane);
      v16bf b = frag_bT(&BT[0][0], 136, ks, nbase, lane);
      acc = wmma_bf16(a, b, acc);
    }
    __syncthreads();
  }

  // epilogue: +bias, q-scale, scatter to head-major [bh][n][64] bf16
  const int col = lane & 15;
  const int rowoff = (lane & 16) ? 8 : 0;
  const int cg = nb + nbase + col;        // 0..1535 (uniform class per subtile)
  const float bv = bias[cg];
  const int which = cg >> 9;              // 0=q 1=k 2=v
  const int hc = cg & 511, h = hc >> 6, d = hc & 63;
  bf16* dst = (which == 0) ? q_ws : (which == 1) ? k_ws : v_ws;
  const float mul = (which == 0) ? QSCALE : 1.0f;
#pragma unroll
  for (int j = 0; j < 8; ++j) {
    const int rg = mb + mbase + j + rowoff;
    const int bi = rg >> 12, ni = rg & 4095;
    dst[(((size_t)(bi * HEADS + h)) * NTOK + ni) * HD + d] =
        (bf16)((acc[j] + bv) * mul);
  }
}

// ---------------- K2: context = k^T v (+ mem kv) ----------------------------
// grid(64) block 256: one block per (b,h); 64x64 f32 accum across 8 waves.
// K/V tiles staged transposed [d][n] (stride 40); LDS double-buffered.
__global__ __launch_bounds__(256)
void k2_context(const bf16* __restrict__ k_ws, const bf16* __restrict__ v_ws,
                const float* __restrict__ mem_kv, bf16* __restrict__ ctx_ws) {
  __shared__ __align__(16) bf16 KT[2][64 * 40];
  __shared__ __align__(16) bf16 VT[2][64 * 40];
  const int t = threadIdx.x, lane = t & 31, wave = t >> 5;
  const int bh = blockIdx.x, h = bh & 7;
  const int d1base = (wave & 3) * 16;
  const int d2base = (wave >> 2) * 32;
  const bf16* kp = k_ws + (size_t)bh * NTOK * HD;
  const bf16* vp = v_ws + (size_t)bh * NTOK * HD;
  v8f acc0 = {}, acc1 = {};

  const int r0 = (t >> 4) * 2;
  const int cb = (t & 15) * 4;
#define STAGE_KV(buf, ni)                                                     \
  {                                                                           \
    v4bf a0 = *(const v4bf*)&kp[(size_t)((ni) + r0) * HD + cb];               \
    v4bf a1 = *(const v4bf*)&kp[(size_t)((ni) + r0 + 1) * HD + cb];           \
    v4bf b0 = *(const v4bf*)&vp[(size_t)((ni) + r0) * HD + cb];               \
    v4bf b1 = *(const v4bf*)&vp[(size_t)((ni) + r0 + 1) * HD + cb];           \
    _Pragma("unroll") for (int i = 0; i < 4; ++i) {                           \
      pack2(&KT[buf][(cb + i) * 40 + r0], a0[i], a1[i]);                      \
      pack2(&VT[buf][(cb + i) * 40 + r0], b0[i], b1[i]);                      \
    }                                                                         \
  }

  STAGE_KV(0, 0);
  __syncthreads();
  for (int ni = 0; ni < NTOK; ni += 32) {
    const int cur = (ni >> 5) & 1;
    if (ni + 32 < NTOK) {
      STAGE_KV(cur ^ 1, ni + 32);
      __builtin_prefetch(&kp[(size_t)(ni + 64 + r0) * HD + cb], 0, 1);
      __builtin_prefetch(&vp[(size_t)(ni + 64 + r0) * HD + cb], 0, 1);
    }
    v16bf a  = frag_a(&KT[cur][0], 40, d1base, 0, lane);   // A = k^T
    v16bf b0 = frag_bT(&VT[cur][0], 40, 0, d2base, lane);
    v16bf b1 = frag_bT(&VT[cur][0], 40, 0, d2base + 16, lane);
    acc0 = wmma_bf16(a, b0, acc0);
    acc1 = wmma_bf16(a, b1, acc1);
    __syncthreads();
  }
#undef STAGE_KV

  // fold in the 4 learned memory kv rows; store context TRANSPOSED
  // (ctx_ws[bh][d2][d1]) so K3 reads it as a contiguous B fragment.
  const int col = lane & 15;
  const int rowoff = (lane & 16) ? 8 : 0;
  const float* mk = mem_kv + (size_t)h * 4 * HD;            // mem_kv[0][h]
  const float* mv = mem_kv + 2048 + (size_t)h * 4 * HD;     // mem_kv[1][h]
#pragma unroll
  for (int j = 0; j < 8; ++j) {
    const int d1 = d1base + j + rowoff;
    float c0 = acc0[j], c1 = acc1[j];
#pragma unroll
    for (int m = 0; m < 4; ++m) {
      const float km = mk[m * HD + d1];
      c0 += km * mv[m * HD + d2base + col];
      c1 += km * mv[m * HD + d2base + 16 + col];
    }
    ctx_ws[(size_t)bh * 4096 + (d2base + col) * 64 + d1]      = (bf16)c0;
    ctx_ws[(size_t)bh * 4096 + (d2base + 16 + col) * 64 + d1] = (bf16)c1;
  }
}

// ---------------- K3: out = q @ ctx ------------------------------------------
// grid(32, 64) block 256: per (b,h), 128(M) x 64(N) tile, K=64.
// Q and ctx tiles are pure bf16 copies -> Tensor Data Mover with LDS padding
// (64-elem rows padded to 72: interval code 4 = 128B, amount code 3 = 16B).
__global__ __launch_bounds__(256)
void k3_apply(const bf16* __restrict__ q_ws, const bf16* __restrict__ ctx_ws,
              bf16* __restrict__ attn_ws) {
  __shared__ __align__(16) bf16 Qs[128][72];     // row-major  [m][k]
  __shared__ __align__(16) bf16 CT[64][72];      // transposed [n][k] (as stored)
  const int t = threadIdx.x, lane = t & 31, wave = t >> 5;
  const int bh = blockIdx.y, nb = blockIdx.x * 128;
  const bf16* qp = q_ws + (size_t)bh * NTOK * HD;

  if (wave == 0) {
    tdm_load_2d_bf16(&Qs[0][0], qp + (size_t)nb * HD,
                     HD, NTOK, HD, 128, HD, 4, 3);
    tdm_load_2d_bf16(&CT[0][0], ctx_ws + (size_t)bh * 4096,
                     HD, HD, HD, HD, HD, 4, 3);
    __builtin_amdgcn_s_wait_tensorcnt(0);
  }
  __syncthreads();

  const int mbase = wave * 16;            // 0..112
  v8f acc[4] = {v8f{}, v8f{}, v8f{}, v8f{}};
#pragma unroll
  for (int kb = 0; kb < 64; kb += 32) {
    v16bf a = frag_a(&Qs[0][0], 72, mbase, kb, lane);   // reused 4x (N tiles)
#pragma unroll
    for (int nt = 0; nt < 4; ++nt) {
      v16bf b = frag_bT(&CT[0][0], 72, kb, nt * 16, lane);
      acc[nt] = wmma_bf16(a, b, acc[nt]);
    }
  }

  const int col = lane & 15;
  const int rowoff = (lane & 16) ? 8 : 0;
  const int bi = bh >> 3, h = bh & 7;
#pragma unroll
  for (int nt = 0; nt < 4; ++nt) {
    const int d = nt * 16 + col;
#pragma unroll
    for (int j = 0; j < 8; ++j) {
      const int ni = nb + mbase + j + rowoff;
      attn_ws[(((size_t)(bi * NTOK + ni)) * HEADS + h) * HD + d] =
          (bf16)acc[nt][j];
    }
  }
}

// ---------------- K4: output projection + scale norm -------------------------
// grid(4, 1024) block 256: 32(M) x 64(N) tile, K=512 staged in 128-chunks.
// A tiles (bf16 copies) go through TDM (128-elem rows = 256B -> interval code
// 5, pad 16B -> stride 136); B tiles need f32->bf16 transpose (VALU path).
__global__ __launch_bounds__(256)
void k4_proj(const bf16* __restrict__ attn_ws, const float* __restrict__ Wo,
             const float* __restrict__ bo, const float* __restrict__ go,
             float* __restrict__ out) {
  __shared__ __align__(16) bf16 As[32][136];     // row-major  [m][k]
  __shared__ __align__(16) bf16 BT[64][136];     // transposed [n][k]
  const int t = threadIdx.x, lane = t & 31, wave = t >> 5;
  const int nb = blockIdx.x * 64;
  const int mb = blockIdx.y * 32;
  const int mbase = (wave >> 2) * 16;
  const int nbase = (wave & 3) * 16;
  v8f acc = {};

#pragma unroll
  for (int kc = 0; kc < 4; ++kc) {
    const int kb = kc * 128;
    if (wave == 0)
      tdm_load_2d_bf16(&As[0][0], &attn_ws[(size_t)mb * HIDDEN + kb],
                       HIDDEN, MROWS, 128, 32, HIDDEN, 5, 3);
    stage_wT(&BT[0][0], Wo, DIM, kb, nb, t);
    if (wave == 0) __builtin_amdgcn_s_wait_tensorcnt(0);
    __syncthreads();
#pragma unroll
    for (int ks = 0; ks < 128; ks += 32) {
      v16bf a = frag_a(&As[0][0], 136, mbase, ks, lane);
      v16bf b = frag_bT(&BT[0][0], 136, ks, nbase, lane);
      acc = wmma_bf16(a, b, acc);
    }
    __syncthreads();
  }

  const int col = lane & 15;
  const int rowoff = (lane & 16) ? 8 : 0;
  const int cg = nb + nbase + col;        // 0..255
  const float bv = bo[cg];
  const float gw = (go[cg] + 1.0f) * RMS_MUL;
#pragma unroll
  for (int j = 0; j < 8; ++j) {
    const int rg = mb + mbase + j + rowoff;
    out[(size_t)rg * DIM + cg] = (acc[j] + bv) * gw;
  }
}

// ---------------------------------------------------------------------------

extern "C" void kernel_launch(void* const* d_in, const int* in_sizes, int n_in,
                              void* d_out, int out_size, void* d_ws,
                              size_t ws_size, hipStream_t stream) {
  const float* x      = (const float*)d_in[0];
  const float* gamma  = (const float*)d_in[1];
  const float* mem_kv = (const float*)d_in[2];
  const float* Wqkv   = (const float*)d_in[3];
  const float* bqkv   = (const float*)d_in[4];
  const float* Wo     = (const float*)d_in[5];
  const float* bo     = (const float*)d_in[6];
  const float* go     = (const float*)d_in[7];
  float* out = (float*)d_out;

  // workspace layout (bf16): q | k | v | ctx ; attn-out reuses k after K2.
  const size_t QKV_BYTES = (size_t)MROWS * HIDDEN * sizeof(bf16);  // 33.5 MB
  char* ws = (char*)d_ws;
  bf16* q_ws    = (bf16*)(ws);
  bf16* k_ws    = (bf16*)(ws + QKV_BYTES);
  bf16* v_ws    = (bf16*)(ws + 2 * QKV_BYTES);
  bf16* ctx_ws  = (bf16*)(ws + 3 * QKV_BYTES);
  bf16* attn_ws = k_ws;   // k dead after k2_context

  k1_qkv<<<dim3(NQKV / 64, MROWS / 32), 256, 0, stream>>>(
      x, gamma, Wqkv, bqkv, q_ws, k_ws, v_ws);
  k2_context<<<dim3(BATCH * HEADS), 256, 0, stream>>>(
      k_ws, v_ws, mem_kv, ctx_ws);
  k3_apply<<<dim3(NTOK / 128, BATCH * HEADS), 256, 0, stream>>>(
      q_ws, ctx_ws, attn_ws);
  k4_proj<<<dim3(DIM / 64, MROWS / 32), 256, 0, stream>>>(
      attn_ws, Wo, bo, go, out);
}
